// VectorQuantizer_39402029974034
// MI455X (gfx1250) — compile-verified, matrix operand order checked by reading, not run on hardware
//
#include <hip/hip_runtime.h>
#include <hip/hip_bf16.h>

// ---------------------------------------------------------------------------
// VectorQuantizer for MI455X (gfx1250, wave32, WMMA + TDM async staging)
//   out = [quantized (16*4096*256 f32)] ++ [indices (16*4096, as f32)] ++ [loss]
// ---------------------------------------------------------------------------

#define D_DIM   256
#define KCODES  4096
#define TOK     65536          // 16 * 4096 tokens
#define NTCH    32             // codes staged in LDS per chunk
#define LSTR    264            // LDS row stride in halves (512B data + 16B pad)
#define MT      2              // M-tiles (of 16 rows) per wave
#define NCHUNK  (KCODES / NTCH)

typedef __attribute__((ext_vector_type(16))) _Float16 v16h;
typedef __attribute__((ext_vector_type(8)))  _Float16 v8h;
typedef __attribute__((ext_vector_type(8)))  float    v8f;
typedef __attribute__((ext_vector_type(4)))  float    v4f;
typedef __attribute__((ext_vector_type(4)))  unsigned int v4u;
typedef __attribute__((ext_vector_type(8)))  int      v8i;
typedef __attribute__((ext_vector_type(4)))  int      v4i;

#if defined(__has_builtin)
#if __has_builtin(__builtin_amdgcn_tensor_load_to_lds) && \
    __has_builtin(__builtin_amdgcn_s_wait_tensorcnt)
#define HAVE_TDM 1
#endif
#endif
#ifndef HAVE_TDM
#define HAVE_TDM 0
#endif

// ---------------------------------------------------------------------------
// Kernel 1: E (f32) -> E (f16) in ws, plus neg_half_e2[k] = -0.5 * ||e_k||^2
// ---------------------------------------------------------------------------
__global__ void vq_prep(const float* __restrict__ emb,
                        _Float16* __restrict__ eh,
                        float* __restrict__ nhe2) {
    const int wave = threadIdx.x >> 5;
    const int lane = threadIdx.x & 31;
    const int row  = blockIdx.x * 8 + wave;

    const float* src = emb + (size_t)row * D_DIM + lane * 8;
    v4f a = *(const v4f*)src;
    v4f b = *(const v4f*)(src + 4);
    v8h h;
    float s = 0.f;
#pragma unroll
    for (int e = 0; e < 4; ++e) { h[e]     = (_Float16)a[e]; s += a[e] * a[e]; }
#pragma unroll
    for (int e = 0; e < 4; ++e) { h[4 + e] = (_Float16)b[e]; s += b[e] * b[e]; }
    *(v8h*)(eh + (size_t)row * D_DIM + lane * 8) = h;

#pragma unroll
    for (int off = 16; off; off >>= 1) s += __shfl_xor(s, off, 32);
    if (lane == 0) nhe2[row] = -0.5f * s;
}

// ---------------------------------------------------------------------------
// TDM: DMA one E chunk (32 codes x 512B) global -> LDS, with HW padding that
// reproduces the LSTR=264-half layout (16B pad after every 512B row).
// ---------------------------------------------------------------------------
#if HAVE_TDM
__device__ __forceinline__ void tdm_stage_chunk(const _Float16* src,
                                                unsigned lds_byte_off) {
    const unsigned long long ga = (unsigned long long)(size_t)src;
    v4u g0;
    g0[0] = 1u;                                   // count=1, user descriptor
    g0[1] = lds_byte_off;                         // LDS destination (bytes)
    g0[2] = (unsigned)(ga & 0xffffffffu);         // global_addr[31:0]
    g0[3] = (unsigned)((ga >> 32) & 0x1ffffffu)   // global_addr[56:32]
            | (2u << 30);                         // type=2 ("image")
    // data_size=4B | pad_enable | pad_interval=6 (128 DW) | pad_amount=3 (4 DW)
    v8i g1;
    g1[0] = (int)0x07920000;
    g1[1] = (int)(128u << 16);                    // tensor_dim0 = 128 DW
    g1[2] = (int)((unsigned)NTCH << 16);          // tensor_dim1 = 32 rows
    g1[3] = (int)(128u << 16);                    // tile_dim0 = 128 DW
    g1[4] = (int)NTCH;                            // tile_dim1 = 32 rows
    g1[5] = 128;                                  // tensor_dim0_stride = 128 DW
    g1[6] = 0;
    g1[7] = 0;
    v4i zz = {0, 0, 0, 0};
#if __clang_major__ >= 23
    v8i z8 = {0, 0, 0, 0, 0, 0, 0, 0};
    __builtin_amdgcn_tensor_load_to_lds(g0, g1, zz, zz, z8, 0);
#else
    __builtin_amdgcn_tensor_load_to_lds(g0, g1, zz, zz, 0);
#endif
}
#endif

// ---------------------------------------------------------------------------
// Per-chunk score GEMM + running argmax update (32 WMMA per call).
// score[m][n] = z_m . e_n - 0.5*||e_n||^2  (argmax <=> argmin of distance)
// ---------------------------------------------------------------------------
__device__ __forceinline__ void compute_chunk(
    const _Float16* __restrict__ sbuf, const float* __restrict__ nhe2,
    const v16h (&Af)[MT][8], float (&bv)[MT][8], int (&bi)[MT][8],
    int nc, int lmod, int lhi) {
    const float nh0 = nhe2[nc + lmod];
    const float nh1 = nhe2[nc + 16 + lmod];
#pragma unroll
    for (int nt = 0; nt < 2; ++nt) {
        const float nh = nt ? nh1 : nh0;
        v8f acc[MT];
#pragma unroll
        for (int t = 0; t < MT; ++t)
#pragma unroll
            for (int e = 0; e < 8; ++e) acc[t][e] = nh;   // bias = -0.5||e||^2

#pragma unroll
        for (int i = 0; i < 8; ++i) {
            // B-frag (column-major B == row of E, same lane pattern as A)
            const _Float16* bp = sbuf + (nt * 16 + lmod) * LSTR + i * 32 + lhi * 8;
            v8h b0 = *(const v8h*)bp;
            v8h b1 = *(const v8h*)(bp + 16);
            v16h Bf = __builtin_shufflevector(b0, b1,
                0, 1, 2, 3, 4, 5, 6, 7, 8, 9, 10, 11, 12, 13, 14, 15);
#pragma unroll
            for (int t = 0; t < MT; ++t)
                acc[t] = __builtin_amdgcn_wmma_f32_16x16x32_f16(
                    false, Af[t][i], false, Bf, (short)0, acc[t], false, false);
        }
        const int col = nc + nt * 16 + lmod;
#pragma unroll
        for (int t = 0; t < MT; ++t)
#pragma unroll
            for (int s = 0; s < 8; ++s)
                if (acc[t][s] > bv[t][s]) { bv[t][s] = acc[t][s]; bi[t][s] = col; }
    }
}

// ---------------------------------------------------------------------------
// Kernel 2: fused score-GEMM + argmin + gather + partial loss
// block = 256 thr (8 waves); wave owns 32 token rows; block owns 256 rows.
// ---------------------------------------------------------------------------
__global__ void __launch_bounds__(256)
vq_main(const float* __restrict__ z, const float* __restrict__ emb,
        const _Float16* __restrict__ eh, const float* __restrict__ nhe2,
        float* __restrict__ out_q, float* __restrict__ out_idx,
        float* __restrict__ partial) {
    __shared__ __align__(16) _Float16 sB[2][NTCH * LSTR];
    __shared__ float sLoss[8];

    const int tid  = threadIdx.x;
    const int wave = tid >> 5;
    const int lane = tid & 31;
    const int lmod = lane & 15;       // row-in-tile (A) / col-in-tile (B)
    const int lhi  = lane >> 4;       // K-phase select
    const int rb   = blockIdx.x * 256 + wave * (MT * 16);

    // ---- Load the wave's A slab: 32 rows x 256 K, f32 -> f16 fragments ----
    v16h Af[MT][8];
#pragma unroll
    for (int t = 0; t < MT; ++t) {
        const int row = rb + t * 16 + lmod;
        const float* zr = z + (size_t)row * D_DIM;
#pragma unroll
        for (int i = 0; i < 8; ++i) {
            const int kb = i * 32 + lhi * 8;
            v4f a0 = *(const v4f*)(zr + kb);
            v4f a1 = *(const v4f*)(zr + kb + 4);
            v4f a2 = *(const v4f*)(zr + kb + 16);
            v4f a3 = *(const v4f*)(zr + kb + 20);
            v16h f;
#pragma unroll
            for (int e = 0; e < 4; ++e) {
                f[e]      = (_Float16)a0[e];
                f[4 + e]  = (_Float16)a1[e];
                f[8 + e]  = (_Float16)a2[e];
                f[12 + e] = (_Float16)a3[e];
            }
            Af[t][i] = f;
        }
    }

    // ---- Running argmax of score (== argmin of distance) ----
    float bv[MT][8];
    int   bi[MT][8];
#pragma unroll
    for (int t = 0; t < MT; ++t)
#pragma unroll
        for (int s = 0; s < 8; ++s) { bv[t][s] = -3.4e38f; bi[t][s] = 0; }

#if HAVE_TDM
    // =========== TDM double-buffered pipeline: copy overlaps WMMA ==========
    const unsigned lds0 = (unsigned)(size_t)(void*)&sB[0][0];
    const unsigned lds1 = (unsigned)(size_t)(void*)&sB[1][0];
    if (wave == 0) {
        tdm_stage_chunk(eh, lds0);                      // chunk 0 -> buf 0
        __builtin_amdgcn_s_wait_tensorcnt(0);
    }
    __syncthreads();
    for (int c = 0; c < NCHUNK; ++c) {
        const int nc = c * NTCH;
        if (wave == 0 && c + 1 < NCHUNK)                // prefetch next chunk
            tdm_stage_chunk(eh + (size_t)(nc + NTCH) * D_DIM,
                            (c & 1) ? lds0 : lds1);
        compute_chunk(sB[c & 1], nhe2, Af, bv, bi, nc, lmod, lhi);
        if (wave == 0) __builtin_amdgcn_s_wait_tensorcnt(0);
        __syncthreads();                                // publish next buffer
    }
#else
    // =================== fallback: cooperative VGPR staging ================
    for (int nc = 0; nc < KCODES; nc += NTCH) {
        const int r  = tid >> 3;
        const int c0 = (tid & 7) * 32;
        const _Float16* src = eh + (size_t)(nc + r) * D_DIM + c0;
        _Float16* dst = &sB[0][0] + r * LSTR + c0;
#pragma unroll
        for (int j = 0; j < 4; ++j)
            *(v8h*)(dst + j * 8) = *(const v8h*)(src + j * 8);
        __syncthreads();
        compute_chunk(&sB[0][0], nhe2, Af, bv, bi, nc, lmod, lhi);
        __syncthreads();
    }
#endif

    // ---- Cross-lane reduce: max score, tie -> smaller index (first argmin) --
#pragma unroll
    for (int off = 1; off < 16; off <<= 1) {
#pragma unroll
        for (int t = 0; t < MT; ++t)
#pragma unroll
            for (int s = 0; s < 8; ++s) {
                float ov = __shfl_xor(bv[t][s], off, 32);
                int   oi = __shfl_xor(bi[t][s], off, 32);
                if (ov > bv[t][s] || (ov == bv[t][s] && oi < bi[t][s])) {
                    bv[t][s] = ov; bi[t][s] = oi;
                }
            }
    }
    // lanes 0-15 agree on rows (t*16 + 0..7); lanes 16-31 on rows (t*16 + 8..15)

    // ---- Write indices, gather exact f32 embedding rows, accumulate loss ----
    float lsum = 0.f;
#pragma unroll
    for (int t = 0; t < MT; ++t) {
#pragma unroll
        for (int r = 0; r < 16; ++r) {
            const int idx  = __shfl(bi[t][r & 7], (r >> 3) * 16, 32);
            const int grow = rb + t * 16 + r;
            if (lane == 0) out_idx[grow] = (float)idx;
            const float* er = emb + (size_t)idx  * D_DIM + lane * 8;
            const float* zr = z   + (size_t)grow * D_DIM + lane * 8;
            float*       qr = out_q + (size_t)grow * D_DIM + lane * 8;
            v4f e0 = *(const v4f*)er;     v4f e1 = *(const v4f*)(er + 4);
            v4f z0 = *(const v4f*)zr;     v4f z1 = *(const v4f*)(zr + 4);
            *(v4f*)qr = e0;               *(v4f*)(qr + 4) = e1;
#pragma unroll
            for (int e = 0; e < 4; ++e) {
                float d0 = e0[e] - z0[e];
                float d1 = e1[e] - z1[e];
                lsum += d0 * d0 + d1 * d1;
            }
        }
    }
#pragma unroll
    for (int off = 16; off; off >>= 1) lsum += __shfl_xor(lsum, off, 32);
    if (lane == 0) sLoss[wave] = lsum;
    __syncthreads();
    if (tid == 0) {
        float s = 0.f;
#pragma unroll
        for (int w = 0; w < 8; ++w) s += sLoss[w];
        partial[blockIdx.x] = s;
    }
}

// ---------------------------------------------------------------------------
// Kernel 3: deterministic final loss reduction over 256 block partials
// ---------------------------------------------------------------------------
__global__ void vq_loss(const float* __restrict__ partial,
                        float* __restrict__ out_loss) {
    __shared__ float sm[256];
    const int tid = threadIdx.x;
    sm[tid] = partial[tid];
    __syncthreads();
    for (int s = 128; s > 0; s >>= 1) {
        if (tid < s) sm[tid] += sm[tid + s];
        __syncthreads();
    }
    if (tid == 0)
        *out_loss = 1.25f * sm[0] / (float)((size_t)TOK * D_DIM);
}

// ---------------------------------------------------------------------------
extern "C" void kernel_launch(void* const* d_in, const int* in_sizes, int n_in,
                              void* d_out, int out_size, void* d_ws, size_t ws_size,
                              hipStream_t stream) {
    const float* z   = (const float*)d_in[0];   // [16,4096,256] f32
    const float* emb = (const float*)d_in[1];   // [4096,256]    f32

    char* ws = (char*)d_ws;
    _Float16* eh   = (_Float16*)ws;                                 // 2 MB
    float*    nhe2 = (float*)(ws + (size_t)KCODES * D_DIM * 2);     // 16 KB
    float*    part = (float*)(ws + (size_t)KCODES * D_DIM * 2 + KCODES * 4);

    float* out      = (float*)d_out;
    float* out_q    = out;                                   // 16,777,216
    float* out_idx  = out + (size_t)TOK * D_DIM;             // 65,536
    float* out_loss = out + (size_t)TOK * D_DIM + TOK;       // 1

    vq_prep<<<KCODES / 8, 256, 0, stream>>>(emb, eh, nhe2);
    vq_main<<<TOK / 256, 256, 0, stream>>>(z, emb, eh, nhe2, out_q, out_idx, part);
    vq_loss<<<1, 256, 0, stream>>>(part, out_loss);
}
